// GNNEncoder_33371895890436
// MI455X (gfx1250) — matrix operand order repaired and checked by reading, not compiled, not run on hardware
//
#include <hip/hip_runtime.h>
#include <hip/hip_bf16.h>

// ---------------------------------------------------------------------------
// GNN encoder (GatedGCN x4) for gfx1250 (MI455X).
// Big GEMMs (131072x128 @ 128x128) run on v_wmma_f32_16x16x32_bf16 with fp32
// accumulation. Each wave keeps its column-tile's weight fragments register-
// resident (4 x v16bf) and sweeps 4 M-tiles against them (16 WMMAs / wave),
// with software double-buffering of A fragments across K-steps.
// ---------------------------------------------------------------------------

#define BB 2
#define NN 256
#define DD 128
#define LL 4
#define RE (BB * NN * NN)   // 131072 edge rows
#define RN (BB * NN)        // 512 node rows

typedef __attribute__((ext_vector_type(16))) __bf16 v16bf;
typedef __attribute__((ext_vector_type(8)))  __bf16 v8bf;
typedef __attribute__((ext_vector_type(8)))  float  v8f;

// ---------------------------------------------------------------------------
// Generic WMMA GEMM: Y(R x 128) = X(bf16, R x 128) @ W + bias, W given as
// Wt (bf16, 128 x 128, transposed: Wt[n][k] = W[k][n]).
// mode 0: Yf = result (+ optional bf16 mirror Ybf)
// mode 1: gate epilogue: Yf = result + Ax[b*N+i, n] + Bx[b*N+j, n]
// mode 2: residual:      Yf += result; Ybf = (bf16)Yf
// One wave -> 4 M-tiles x 1 N-tile (64x16 output). R must be a multiple of 64.
// ---------------------------------------------------------------------------
__global__ __launch_bounds__(128) void wmma_gemm128(
    const __bf16* __restrict__ X, const __bf16* __restrict__ Wt,
    const float* __restrict__ bias, float* __restrict__ Yf,
    __bf16* __restrict__ Ybf, const float* __restrict__ Ax,
    const float* __restrict__ Bx, int R, int mode)
{
    const int wave = threadIdx.x >> 5;
    const int lane = threadIdx.x & 31;
    const int tile = blockIdx.x * 4 + wave;
    const int nSup = R >> 6;      // 64 rows per super-tile
    const int mt   = tile >> 3;   // 128/16 = 8 column tiles
    const int nt   = tile & 7;
    if (mt >= nSup) return;

    const int row0 = mt << 6;
    const int col0 = nt << 4;
    const int lr   = lane & 15;
    const int kA   = (lane < 16) ? 0 : 16;   // A: K halves {0..15}/{16..31}
    const int kB   = (lane < 16) ? 0 : 16;   // B: K halves {0..15}/{16..31}

    // ---- register-resident weight fragments for this column tile ----
    const __bf16* __restrict__ Wrow = Wt + (size_t)(col0 + lr) * DD + kB;
    v16bf bfrag[4];
#pragma unroll
    for (int kb = 0; kb < 4; ++kb)
        bfrag[kb] = *(const v16bf*)(Wrow + kb * 32);

    // ---- A fragments: double-buffered across K-steps ----
    const __bf16* __restrict__ Xbase = X + (size_t)(row0 + lr) * DD + kA;
    auto loadA = [&](int m, int kb) -> v16bf {
        const __bf16* p = Xbase + (size_t)(m * 16) * DD + kb * 32;
        v8bf lo = *(const v8bf*)(p);
        v8bf hi = *(const v8bf*)(p + 8);
        v16bf a;
#pragma unroll
        for (int i = 0; i < 8; ++i) { a[i] = lo[i]; a[8 + i] = hi[i]; }
        return a;
    };

    v16bf af[2][4];
#pragma unroll
    for (int m = 0; m < 4; ++m) af[0][m] = loadA(m, 0);

    v8f acc[4] = {{}, {}, {}, {}};
#pragma unroll
    for (int kb = 0; kb < 4; ++kb) {
        if (kb < 3) {
#pragma unroll
            for (int m = 0; m < 4; ++m)
                af[(kb + 1) & 1][m] = loadA(m, kb + 1);   // prefetch next K-step
        }
#pragma unroll
        for (int m = 0; m < 4; ++m)
            acc[m] = __builtin_amdgcn_wmma_f32_16x16x32_bf16(
                         false, af[kb & 1][m], false, bfrag[kb], (short)0,
                         acc[m], false, false);
    }

    const int  madd = (lane < 16) ? 0 : 8;
    const int  ncol = col0 + lr;
    const float bc  = bias ? bias[ncol] : 0.0f;

#pragma unroll
    for (int m = 0; m < 4; ++m) {
        const int rbase = row0 + m * 16;
        if (mode == 1) {
            // A 16-row tile never crosses a (b,i) boundary (256 % 16 == 0).
            const int bi = rbase >> 8;         // b*N + i
            const int b_ = bi >> 8;
            const float axv = Ax[(size_t)bi * DD + ncol];
#pragma unroll
            for (int v = 0; v < 8; ++v) {
                const int r = rbase + madd + v;
                const int j = r & 255;
                float val = acc[m][v] + bc + axv
                          + Bx[((size_t)(b_ << 8) + j) * DD + ncol];
                Yf[(size_t)r * DD + ncol] = val;
            }
        } else if (mode == 2) {
#pragma unroll
            for (int v = 0; v < 8; ++v) {
                const size_t idx = (size_t)(rbase + madd + v) * DD + ncol;
                float ev = Yf[idx] + acc[m][v] + bc;
                Yf[idx]  = ev;
                Ybf[idx] = (__bf16)ev;
            }
        } else {
#pragma unroll
            for (int v = 0; v < 8; ++v) {
                const size_t idx = (size_t)(rbase + madd + v) * DD + ncol;
                float val = acc[m][v] + bc;
                Yf[idx] = val;
                if (Ybf) Ybf[idx] = (__bf16)val;
            }
        }
    }
}

// ---- transpose + fp32->bf16 convert for one 128x128 weight matrix ----------
__global__ __launch_bounds__(128) void conv_wt(const float* __restrict__ W,
                                               __bf16* __restrict__ Wt)
{
    const int k = blockIdx.x;
    const int n = threadIdx.x;
    Wt[(size_t)n * DD + k] = (__bf16)W[(size_t)k * DD + n];
}

// ---- block-wide (128 thread) reductions -----------------------------------
__device__ __forceinline__ void blk_reduce2(float* r1, float* r2, int d)
{
    __syncthreads();
#pragma unroll
    for (int s = 64; s > 0; s >>= 1) {
        if (d < s) { r1[d] += r1[d + s]; r2[d] += r2[d + s]; }
        __syncthreads();
    }
}

// ---- node positional embedding (sine) -> bf16 rows ------------------------
__global__ __launch_bounds__(128) void node_embed(const float* __restrict__ coords,
                                                  __bf16* __restrict__ P)
{
    const int r = blockIdx.x;           // b*N + i
    const int d = threadIdx.x;          // 0..127
    const float c  = coords[r * 2 + (d >> 6)];   // d<64: y, else x
    const int   k  = d & 63;
    const int   p  = k >> 1;
    // dim_t = 10000^(p/32) ; scale = 2*pi
    const float inv = __expf(-(float)p * (9.2103403719761836f / 32.0f));
    const float u   = c * 6.2831853071795864f * inv;
    const float val = (k & 1) ? __cosf(u) : __sinf(u);
    P[(size_t)r * DD + d] = (__bf16)val;
}

// ---- edge scalar embedding (sine) -> bf16 rows ----------------------------
__global__ __launch_bounds__(128) void edge_embed(const float* __restrict__ adj,
                                                  __bf16* __restrict__ Epos)
{
    const size_t r = blockIdx.x;        // edge row
    const int    d = threadIdx.x;
    const float  a = adj[r];
    const int    p = d >> 1;
    // dim_t = 10000^(p/64)
    const float inv = __expf(-(float)p * (9.2103403719761836f / 64.0f));
    const float u   = a * inv;
    const float val = (d & 1) ? __cosf(u) : __sinf(u);
    Epos[r * DD + d] = (__bf16)val;
}

// ---- timestep MLP + per-layer time projections (tiny, single block) -------
__global__ __launch_bounds__(128) void time_embed(
    const float* __restrict__ t,
    const float* __restrict__ te1_W, const float* __restrict__ te1_b,
    const float* __restrict__ te2_W, const float* __restrict__ te2_b,
    const float* __restrict__ tl_W,  const float* __restrict__ tl_b,
    float* __restrict__ tproj)        // [L][B][128]
{
    const int tid = threadIdx.x;
    __shared__ float temb[BB][128], h1[BB][64], rte[BB][64];
    for (int b = 0; b < BB; ++b) {
        const int k   = tid & 63;
        const float f = __expf(-(float)k * (9.2103403719761836f / 64.0f));
        const float a = t[b] * f;
        temb[b][tid]  = (tid < 64) ? __cosf(a) : __sinf(a);
    }
    __syncthreads();
    if (tid < 64)
        for (int b = 0; b < BB; ++b) {
            float acc = te1_b[tid];
            for (int d = 0; d < 128; ++d) acc += temb[b][d] * te1_W[d * 64 + tid];
            h1[b][tid] = fmaxf(acc, 0.0f);
        }
    __syncthreads();
    if (tid < 64)
        for (int b = 0; b < BB; ++b) {
            float acc = te2_b[tid];
            for (int k = 0; k < 64; ++k) acc += h1[b][k] * te2_W[k * 64 + tid];
            rte[b][tid] = fmaxf(acc, 0.0f);   // relu(te) used by every layer
        }
    __syncthreads();
    for (int l = 0; l < LL; ++l)
        for (int b = 0; b < BB; ++b) {
            float acc = tl_b[l * 128 + tid];
            for (int k = 0; k < 64; ++k) acc += rte[b][k] * tl_W[(l * 64 + k) * 128 + tid];
            tproj[(l * BB + b) * 128 + tid] = acc;
        }
}

// ---- agg[b,i,d] = sum_j sigmoid(gate[b,i,j,d]) * Vh[b,j,d] ----------------
__global__ __launch_bounds__(128) void gate_agg(
    const float* __restrict__ gate, const float* __restrict__ Vh,
    float* __restrict__ agg)
{
    const int bi = blockIdx.x;          // b*N + i
    const int d  = threadIdx.x;
    const int b  = bi >> 8;
    const float* __restrict__ grow = gate + (size_t)bi * NN * DD + d;
    const float* __restrict__ vrow = Vh + (size_t)(b << 8) * DD + d;
    float acc = 0.0f;
    for (int j = 0; j < NN; ++j) {
        if (j + 8 < NN) __builtin_prefetch(grow + (size_t)(j + 8) * DD, 0, 0);
        const float g = grow[(size_t)j * DD];
        acc += vrow[(size_t)j * DD] / (1.0f + __expf(-g));   // sigmoid(g)*Vh
    }
    agg[(size_t)bi * DD + d] = acc;
}

// ---- x = x + relu(LN(Uh + agg)); keep bf16 mirror -------------------------
__global__ __launch_bounds__(128) void node_update(
    const float* __restrict__ Uh, const float* __restrict__ agg,
    const float* __restrict__ lnh_s, const float* __restrict__ lnh_b,
    float* __restrict__ x, __bf16* __restrict__ xbf)
{
    const int row = blockIdx.x;
    const int d   = threadIdx.x;
    const size_t idx = (size_t)row * DD + d;
    const float v = Uh[idx] + agg[idx];
    __shared__ float r1[128], r2[128];
    r1[d] = v; r2[d] = v * v;
    blk_reduce2(r1, r2, d);
    const float mu  = r1[0] * (1.0f / 128.0f);
    const float var = r2[0] * (1.0f / 128.0f) - mu * mu;
    float h = (v - mu) * rsqrtf(var + 1e-5f) * lnh_s[d] + lnh_b[d];
    h = fmaxf(h, 0.0f);
    const float xn = x[idx] + h;
    x[idx]   = xn;
    xbf[idx] = (__bf16)xn;
}

// ---- edge epilogue: S = bf16( silu( LN2( relu(LN1(gate)) + tproj ) ) ) ----
__global__ __launch_bounds__(128) void edge_update(
    const float* __restrict__ gate, const float* __restrict__ tproj, // [B][128]
    const float* __restrict__ lne_s, const float* __restrict__ lne_b,
    const float* __restrict__ pln_s, const float* __restrict__ pln_b,
    __bf16* __restrict__ S)
{
    const int r = blockIdx.x;
    const int d = threadIdx.x;
    const int b = r >> 16;              // r / (N*N)
    const size_t idx = (size_t)r * DD + d;
    const float v = gate[idx];
    __shared__ float r1[128], r2[128];
    r1[d] = v; r2[d] = v * v;
    blk_reduce2(r1, r2, d);
    float mu  = r1[0] * (1.0f / 128.0f);
    float var = r2[0] * (1.0f / 128.0f) - mu * mu;
    float h = fmaxf((v - mu) * rsqrtf(var + 1e-5f) * lne_s[d] + lne_b[d], 0.0f);
    h += tproj[b * 128 + d];
    __syncthreads();                    // protect r1[0]/r2[0] before reuse
    r1[d] = h; r2[d] = h * h;
    blk_reduce2(r1, r2, d);
    mu  = r1[0] * (1.0f / 128.0f);
    var = r2[0] * (1.0f / 128.0f) - mu * mu;
    const float h2 = (h - mu) * rsqrtf(var + 1e-5f) * pln_s[d] + pln_b[d];
    const float s  = h2 / (1.0f + __expf(-h2));   // silu
    S[idx] = (__bf16)s;
}

// ---- GroupNorm stats: per (b, group) sum / sumsq via atomics --------------
__global__ __launch_bounds__(128) void gn_stats(const float* __restrict__ e,
                                                float* __restrict__ stats)
{
    const int bi = blockIdx.x;          // b*N + i
    const int d  = threadIdx.x;
    const int b  = bi >> 8;
    const float* __restrict__ row = e + (size_t)bi * NN * DD + d;
    float s = 0.0f, s2 = 0.0f;
    for (int j = 0; j < NN; ++j) {
        const float v = row[(size_t)j * DD];
        s += v; s2 += v * v;
    }
    __shared__ float r1[128], r2[128];
    r1[d] = s; r2[d] = s2;
    __syncthreads();
    if ((d & 3) == 0) {
        const float gs  = r1[d] + r1[d + 1] + r1[d + 2] + r1[d + 3];
        const float gs2 = r2[d] + r2[d + 1] + r2[d + 2] + r2[d + 3];
        const int g = d >> 2;
        atomicAdd(&stats[(b * 32 + g) * 2 + 0], gs);
        atomicAdd(&stats[(b * 32 + g) * 2 + 1], gs2);
    }
}

// ---- out[b,0,i,j] = sum_d relu(GN(e)) * conv_W[d] + conv_b ----------------
__global__ __launch_bounds__(128) void gn_out(
    const float* __restrict__ e, const float* __restrict__ stats,
    const float* __restrict__ gn_s, const float* __restrict__ gn_b,
    const float* __restrict__ conv_W, const float* __restrict__ conv_b,
    float* __restrict__ out)
{
    const int r = blockIdx.x;
    const int d = threadIdx.x;
    const int b = r >> 16;
    const int g = d >> 2;
    const float cnt = 4.0f * (float)NN * (float)NN;
    const float su = stats[(b * 32 + g) * 2 + 0];
    const float sq = stats[(b * 32 + g) * 2 + 1];
    const float mu  = su / cnt;
    const float var = sq / cnt - mu * mu;
    const float v   = e[(size_t)r * DD + d];
    float xn = (v - mu) * rsqrtf(var + 1e-5f) * gn_s[d] + gn_b[d];
    xn = fmaxf(xn, 0.0f) * conv_W[d];
    __shared__ float r1[128];
    r1[d] = xn;
    __syncthreads();
#pragma unroll
    for (int s = 64; s > 0; s >>= 1) {
        if (d < s) r1[d] += r1[d + s];
        __syncthreads();
    }
    if (d == 0) out[r] = r1[0] + conv_b[0];
}

__global__ void zero_buf(float* p, int n)
{
    int i = blockIdx.x * blockDim.x + threadIdx.x;
    if (i < n) p[i] = 0.0f;
}

// ---------------------------------------------------------------------------
extern "C" void kernel_launch(void* const* d_in, const int* in_sizes, int n_in,
                              void* d_out, int out_size, void* d_ws, size_t ws_size,
                              hipStream_t stream)
{
    (void)in_sizes; (void)n_in; (void)out_size; (void)ws_size;

    // ---- inputs (setup_inputs order) ----
    const float* coords = (const float*)d_in[0];
    const float* adj_t  = (const float*)d_in[1];
    const float* t_in   = (const float*)d_in[2];
    const float* node_W = (const float*)d_in[3];
    const float* node_b = (const float*)d_in[4];
    const float* edge_W = (const float*)d_in[5];
    const float* edge_b = (const float*)d_in[6];
    const float* te1_W  = (const float*)d_in[7];
    const float* te1_b  = (const float*)d_in[8];
    const float* te2_W  = (const float*)d_in[9];
    const float* te2_b  = (const float*)d_in[10];
    const float* A_W    = (const float*)d_in[11];
    const float* A_b    = (const float*)d_in[12];
    const float* B_W    = (const float*)d_in[13];
    const float* B_b    = (const float*)d_in[14];
    const float* C_W    = (const float*)d_in[15];
    const float* C_b    = (const float*)d_in[16];
    const float* U_W    = (const float*)d_in[17];
    const float* U_b    = (const float*)d_in[18];
    const float* V_W    = (const float*)d_in[19];
    const float* V_b    = (const float*)d_in[20];
    const float* lnh_s  = (const float*)d_in[21];
    const float* lnh_b  = (const float*)d_in[22];
    const float* lne_s  = (const float*)d_in[23];
    const float* lne_b  = (const float*)d_in[24];
    const float* tl_W   = (const float*)d_in[25];
    const float* tl_b   = (const float*)d_in[26];
    const float* pln_s  = (const float*)d_in[27];
    const float* pln_b  = (const float*)d_in[28];
    const float* plo_W  = (const float*)d_in[29];
    const float* plo_b  = (const float*)d_in[30];
    const float* gn_s   = (const float*)d_in[31];
    const float* gn_b   = (const float*)d_in[32];
    const float* conv_W = (const float*)d_in[33];
    const float* conv_b = (const float*)d_in[34];
    float* out = (float*)d_out;

    // ---- workspace layout ----
    char* ws = (char*)d_ws;
    size_t off = 0;
    auto alloc = [&](size_t bytes) { size_t o = off; off += (bytes + 255) & ~(size_t)255; return o; };
    float*  e_f32 = (float*)(ws + alloc((size_t)RE * DD * 4));
    float*  gate  = (float*)(ws + alloc((size_t)RE * DD * 4));
    __bf16* e_bf  = (__bf16*)(ws + alloc((size_t)RE * DD * 2));
    __bf16* S_bf  = (__bf16*)(ws + alloc((size_t)RE * DD * 2));  // also Epos staging
    float*  x_f32 = (float*)(ws + alloc((size_t)RN * DD * 4));
    __bf16* x_bf  = (__bf16*)(ws + alloc((size_t)RN * DD * 2));
    __bf16* P_bf  = (__bf16*)(ws + alloc((size_t)RN * DD * 2));
    float*  Ax    = (float*)(ws + alloc((size_t)RN * DD * 4));
    float*  Bx    = (float*)(ws + alloc((size_t)RN * DD * 4));
    float*  Uh    = (float*)(ws + alloc((size_t)RN * DD * 4));
    float*  Vh    = (float*)(ws + alloc((size_t)RN * DD * 4));
    float*  agg   = (float*)(ws + alloc((size_t)RN * DD * 4));
    float*  tproj = (float*)(ws + alloc((size_t)LL * BB * DD * 4));
    __bf16* wt    = (__bf16*)(ws + alloc((size_t)26 * DD * DD * 2)); // bf16 W^T pool
    float*  stats = (float*)(ws + alloc((size_t)BB * 32 * 2 * 4));

    const size_t WSZ = (size_t)DD * DD;   // elements per 128x128 matrix
    auto wtp = [&](int i) { return wt + (size_t)i * WSZ; };

    // ---- 0) zero GN stats (must happen every launch: graph replays) ----
    zero_buf<<<1, 128, 0, stream>>>(stats, BB * 32 * 2);

    // ---- 1) weight transpose + bf16 convert (26 matrices) ----
    conv_wt<<<DD, DD, 0, stream>>>(node_W, wtp(0));
    conv_wt<<<DD, DD, 0, stream>>>(edge_W, wtp(1));
    for (int l = 0; l < LL; ++l) {
        conv_wt<<<DD, DD, 0, stream>>>(A_W   + (size_t)l * WSZ, wtp(2 + l * 6 + 0));
        conv_wt<<<DD, DD, 0, stream>>>(B_W   + (size_t)l * WSZ, wtp(2 + l * 6 + 1));
        conv_wt<<<DD, DD, 0, stream>>>(C_W   + (size_t)l * WSZ, wtp(2 + l * 6 + 2));
        conv_wt<<<DD, DD, 0, stream>>>(U_W   + (size_t)l * WSZ, wtp(2 + l * 6 + 3));
        conv_wt<<<DD, DD, 0, stream>>>(V_W   + (size_t)l * WSZ, wtp(2 + l * 6 + 4));
        conv_wt<<<DD, DD, 0, stream>>>(plo_W + (size_t)l * WSZ, wtp(2 + l * 6 + 5));
    }

    // ---- 2) embeddings ----
    node_embed<<<RN, DD, 0, stream>>>(coords, P_bf);
    edge_embed<<<RE, DD, 0, stream>>>(adj_t, S_bf);   // S_bf temporarily = Epos
    time_embed<<<1, 128, 0, stream>>>(t_in, te1_W, te1_b, te2_W, te2_b, tl_W, tl_b, tproj);

    // Grid: (R/64 super-tiles) * 8 column tiles, 4 waves (tiles) per block.
    const int gN = (RN / 64) * 8 / 4;    // node-sized GEMMs (R=512)  -> 16 blocks
    const int gE = (RE / 64) * 8 / 4;    // edge-sized GEMMs (R=131072) -> 4096 blocks
    wmma_gemm128<<<gN, 128, 0, stream>>>(P_bf, wtp(0), node_b, x_f32, x_bf,
                                         nullptr, nullptr, RN, 0);
    wmma_gemm128<<<gE, 128, 0, stream>>>(S_bf, wtp(1), edge_b, e_f32, e_bf,
                                         nullptr, nullptr, RE, 0);

    // ---- 3) GatedGCN layers ----
    for (int l = 0; l < LL; ++l) {
        const float* Ab = A_b + l * DD;  const float* Bb = B_b + l * DD;
        const float* Cb = C_b + l * DD;  const float* Ub = U_b + l * DD;
        const float* Vb = V_b + l * DD;  const float* Pb = plo_b + l * DD;

        wmma_gemm128<<<gN, 128, 0, stream>>>(x_bf, wtp(2 + l * 6 + 0), Ab, Ax,
                                             nullptr, nullptr, nullptr, RN, 0);
        wmma_gemm128<<<gN, 128, 0, stream>>>(x_bf, wtp(2 + l * 6 + 1), Bb, Bx,
                                             nullptr, nullptr, nullptr, RN, 0);
        wmma_gemm128<<<gN, 128, 0, stream>>>(x_bf, wtp(2 + l * 6 + 3), Ub, Uh,
                                             nullptr, nullptr, nullptr, RN, 0);
        wmma_gemm128<<<gN, 128, 0, stream>>>(x_bf, wtp(2 + l * 6 + 4), Vb, Vh,
                                             nullptr, nullptr, nullptr, RN, 0);

        // gate = e @ C_W + C_b + Ax[i] + Bx[j]   (fused epilogue)
        wmma_gemm128<<<gE, 128, 0, stream>>>(e_bf, wtp(2 + l * 6 + 2), Cb, gate,
                                             nullptr, Ax, Bx, RE, 1);

        gate_agg<<<RN, DD, 0, stream>>>(gate, Vh, agg);
        node_update<<<RN, DD, 0, stream>>>(Uh, agg, lnh_s + l * DD, lnh_b + l * DD,
                                           x_f32, x_bf);
        edge_update<<<RE, DD, 0, stream>>>(gate, tproj + l * BB * DD,
                                           lne_s + l * DD, lne_b + l * DD,
                                           pln_s + l * DD, pln_b + l * DD, S_bf);

        // e += S @ plo_W + plo_b  (fused residual + bf16 mirror)
        wmma_gemm128<<<gE, 128, 0, stream>>>(S_bf, wtp(2 + l * 6 + 5), Pb, e_f32,
                                             e_bf, nullptr, nullptr, RE, 2);
    }

    // ---- 4) GroupNorm(32) + ReLU + 1x1 conv (D -> 1) ----
    gn_stats<<<RN, DD, 0, stream>>>(e_f32, stats);
    gn_out<<<RE, DD, 0, stream>>>(e_f32, stats, gn_s, gn_b, conv_W, conv_b, out);
}
